// FlaxBasicTransformerBlock2_53300544143374
// MI455X (gfx1250) — matrix operand order repaired and checked by reading, not compiled, not run on hardware
//
#include <hip/hip_runtime.h>

// ---------------------------------------------------------------------------
// Problem constants (from reference)
// ---------------------------------------------------------------------------
#define BATCH   2
#define NTOK    1024          // H*W = 32*32
#define CDIM    512
#define EDIM    768
#define SENC    77
#define SENC_P  80            // padded to multiple of 16
#define HEADS   64            // split_heads: 64 heads of dim 8 (faithful quirk)
#define DHEAD   8
#define LTOT    1101          // 77 + 1024
#define LPAD    1120          // multiple of 32 for WMMA K-steps
#define GROUPS  32
#define CG      16            // CDIM / GROUPS

typedef __attribute__((ext_vector_type(16))) __bf16 v16bf;
typedef __attribute__((ext_vector_type(8)))  __bf16 v8bf;
typedef __attribute__((ext_vector_type(8)))  float  v8f;

// WMMA 16x16x32 bf16 fragment layouts (wave32, CDNA5 ISA 7.12.2):
//  A: lane = hi*16 + m ; per-lane elements = two contiguous 8-elem K-runs:
//     K in [hi*8, hi*8+8) then [16+hi*8, 16+hi*8+8)
//  B: lane = hi*16 + n ; per-lane elements = two contiguous 8-elem K-runs:
//     K in [hi*8, hi*8+8) then [16+hi*8, 16+hi*8+8)
__device__ __forceinline__ v16bf load_a_frag(const __bf16* base, int hi) {
  v8bf lo = *(const v8bf*)(base + hi * 8);
  v8bf hi8 = *(const v8bf*)(base + 16 + hi * 8);
  v16bf f;
#pragma unroll
  for (int e = 0; e < 8; ++e) { f[e] = lo[e]; f[8 + e] = hi8[e]; }
  return f;
}
__device__ __forceinline__ v16bf load_b_frag(const __bf16* base, int hi) {
  v8bf lo = *(const v8bf*)(base + hi * 8);
  v8bf hi8 = *(const v8bf*)(base + 16 + hi * 8);
  v16bf f;
#pragma unroll
  for (int e = 0; e < 8; ++e) { f[e] = lo[e]; f[8 + e] = hi8[e]; }
  return f;
}

// ---------------------------------------------------------------------------
// LayerNorm over last dim (768) -> bf16, rows >= 77 zero-padded to SENC_P
// grid (SENC_P, B), block 256
// ---------------------------------------------------------------------------
__global__ __launch_bounds__(256)
void layernorm_bf16_k(const float* __restrict__ x, const float* __restrict__ scale,
                      const float* __restrict__ bias, __bf16* __restrict__ y) {
  const int s = blockIdx.x, b = blockIdx.y, t = threadIdx.x;
  __bf16* yr = y + ((long)b * SENC_P + s) * EDIM;
  if (s >= SENC) {
    for (int i = t; i < EDIM; i += 256) yr[i] = (__bf16)0.f;
    return;
  }
  const float* xr = x + ((long)b * SENC + s) * EDIM;
  __shared__ float red[2][256];
  float sum = 0.f, sq = 0.f;
  for (int i = t; i < EDIM; i += 256) { float v = xr[i]; sum += v; sq += v * v; }
  red[0][t] = sum; red[1][t] = sq;
  __syncthreads();
  for (int w = 128; w > 0; w >>= 1) {
    if (t < w) { red[0][t] += red[0][t + w]; red[1][t] += red[1][t + w]; }
    __syncthreads();
  }
  const float mu  = red[0][0] * (1.f / EDIM);
  const float var = red[1][0] * (1.f / EDIM) - mu * mu;
  const float rstd = rsqrtf(var + 1e-5f);
  for (int i = t; i < EDIM; i += 256)
    yr[i] = (__bf16)((xr[i] - mu) * rstd * scale[i] + bias[i]);
}

// ---------------------------------------------------------------------------
// GroupNorm: stats per (b, group) over (N=1024, CG=16) -> bf16
// grid (GROUPS, B), block 256
// ---------------------------------------------------------------------------
__global__ __launch_bounds__(256)
void groupnorm_bf16_k(const float* __restrict__ x, const float* __restrict__ scale,
                      const float* __restrict__ bias, __bf16* __restrict__ y) {
  const int g = blockIdx.x, b = blockIdx.y, t = threadIdx.x;
  const float* xb = x + (long)b * NTOK * CDIM + g * CG;
  __bf16*      yb = y + (long)b * NTOK * CDIM + g * CG;
  __shared__ float red[2][256];
  float s = 0.f, s2 = 0.f;
  for (int i = t; i < NTOK * CG; i += 256) {
    int n = i / CG, c = i % CG;
    float v = xb[(long)n * CDIM + c];
    s += v; s2 += v * v;
  }
  red[0][t] = s; red[1][t] = s2;
  __syncthreads();
  for (int w = 128; w > 0; w >>= 1) {
    if (t < w) { red[0][t] += red[0][t + w]; red[1][t] += red[1][t + w]; }
    __syncthreads();
  }
  const float inv = 1.f / (NTOK * CG);
  const float mu  = red[0][0] * inv;
  const float var = red[1][0] * inv - mu * mu;
  const float rstd = rsqrtf(var + 1e-5f);
  for (int i = t; i < NTOK * CG; i += 256) {
    int n = i / CG, c = i % CG;
    int cc = g * CG + c;
    float v = (xb[(long)n * CDIM + c] - mu) * rstd;
    yb[(long)n * CDIM + c] = (__bf16)(v * scale[cc] + bias[cc]);
  }
}

// ---------------------------------------------------------------------------
// Convert fp32 weight [K, Nc] -> bf16 transposed [Nc, K] (contiguous B-frag loads)
// ---------------------------------------------------------------------------
__global__ void convert_w_t_k(const float* __restrict__ W, __bf16* __restrict__ Wt,
                              int K, int Nc) {
  long idx = (long)blockIdx.x * blockDim.x + threadIdx.x;
  if (idx >= (long)K * Nc) return;
  int k = (int)(idx / Nc), n = (int)(idx % Nc);
  Wt[(long)n * K + k] = (__bf16)W[idx];
}

__global__ void zero_bf16_k(__bf16* __restrict__ p, long n) {
  long i = (long)blockIdx.x * blockDim.x + threadIdx.x;
  if (i < n) p[i] = (__bf16)0.f;
}

// ---------------------------------------------------------------------------
// bf16 WMMA GEMM, register-blocked 16(M) x 64(N) per wave:
//   one A fragment shared across 4 B fragments -> 4 WMMAs / k-step.
//   A  : bf16 [B, *, lda], batch stride aBatch
//   Wt : bf16 [Nc, K] (transposed weight)
//   out: optional f32 and/or bf16, ld = ldc, batch stride cBatch
// grid (Nc/64, ceil(M/16), B), block 32.
// ---------------------------------------------------------------------------
__global__ __launch_bounds__(32)
void gemm_bf16_wmma_k(const __bf16* __restrict__ A, int lda, long aBatch,
                      const __bf16* __restrict__ Wt, int K,
                      const float* __restrict__ bias,
                      float* __restrict__ Cf, __bf16* __restrict__ Cb,
                      int ldc, long cBatch, int rowOff, int Mvalid) {
  const int lane = threadIdx.x;
  const int mrow = lane & 15, hi = lane >> 4;
  const int m0 = blockIdx.y * 16, n0 = blockIdx.x * 64, b = blockIdx.z;
  const __bf16* Ap = A + (long)b * aBatch + (long)(m0 + mrow) * lda;
  const __bf16* Bp0 = Wt + (long)(n0 + mrow) * K;
  const __bf16* Bp1 = Bp0 + (long)16 * K;
  const __bf16* Bp2 = Bp0 + (long)32 * K;
  const __bf16* Bp3 = Bp0 + (long)48 * K;
  v8f acc0 = {}, acc1 = {}, acc2 = {}, acc3 = {};
  for (int k0 = 0; k0 < K; k0 += 32) {
    __builtin_prefetch(Ap + k0 + 128, 0, 0);    // global_prefetch_b8
    __builtin_prefetch(Bp0 + k0 + 128, 0, 0);
    v16bf af = load_a_frag(Ap + k0, hi);
    v16bf b0 = load_b_frag(Bp0 + k0, hi);
    v16bf b1 = load_b_frag(Bp1 + k0, hi);
    v16bf b2 = load_b_frag(Bp2 + k0, hi);
    v16bf b3 = load_b_frag(Bp3 + k0, hi);
    acc0 = __builtin_amdgcn_wmma_f32_16x16x32_bf16(false, af, false, b0, (short)0, acc0, false, false);
    acc1 = __builtin_amdgcn_wmma_f32_16x16x32_bf16(false, af, false, b1, (short)0, acc1, false, false);
    acc2 = __builtin_amdgcn_wmma_f32_16x16x32_bf16(false, af, false, b2, (short)0, acc2, false, false);
    acc3 = __builtin_amdgcn_wmma_f32_16x16x32_bf16(false, af, false, b3, (short)0, acc3, false, false);
  }
  v8f accs[4] = {acc0, acc1, acc2, acc3};
#pragma unroll
  for (int t = 0; t < 4; ++t) {
    const int col = n0 + t * 16 + mrow;
    const float bv = bias ? bias[col] : 0.f;
#pragma unroll
    for (int r = 0; r < 8; ++r) {
      int m = m0 + r + (hi ? 8 : 0);
      if (m < Mvalid) {
        float v = accs[t][r] + bv;
        long idx = (long)b * cBatch + (long)(m + rowOff) * ldc + col;
        if (Cf) Cf[idx] = v;
        if (Cb) Cb[idx] = (__bf16)v;
      }
    }
  }
}

// ---------------------------------------------------------------------------
// Attention: per (query-tile of 16, head, batch). 64 heads of dim 8.
//   scores = (Q Kt) * 1/sqrt(8)  via bf16 WMMA (head dim padded 8->32)
//   softmax over LPAD cols (padded cols masked to -1e30 -> prob 0)
//   out = P @ V via bf16 WMMA (dim-8 output padded to 16 cols)
// V tile (1120 x 8 bf16 = 17.5 KB) staged into LDS with coalesced 16B loads.
// block = 1 wave (32), grid (NTOK/16, HEADS, B). Static LDS = 52.5 KB.
// ---------------------------------------------------------------------------
__global__ __launch_bounds__(32)
void attn_wmma_k(const __bf16* __restrict__ Q, const __bf16* __restrict__ Kb,
                 const __bf16* __restrict__ Vb, __bf16* __restrict__ O) {
  __shared__ alignas(16) __bf16 sc[16 * LPAD];      // 35 KB scores/probs
  __shared__ alignas(16) __bf16 vlds[LPAD * DHEAD]; // 17.5 KB V tile
  const int lane = threadIdx.x, mrow = lane & 15, hi = lane >> 4;
  const int qt = blockIdx.x, h = blockIdx.y, b = blockIdx.z;
  const float scale = 0.3535533905932738f;   // 1/sqrt(8)

  // stage V[:, head] into LDS: one 16B row per lane per step
  for (int j = lane; j < LPAD; j += 32) {
    *(v8bf*)(vlds + j * DHEAD) =
        *(const v8bf*)(Vb + ((long)b * LPAD + j) * CDIM + h * DHEAD);
  }

  // A fragment: Q rows (head dim padded 8 -> 32 with zeros; only hi==0 lanes
  // carry real K values, per the A layout's K = hi*8 + e mapping)
  v16bf aq = {};
  if (hi == 0) {
    v8bf q8 = *(const v8bf*)(Q + ((long)b * NTOK + qt * 16 + mrow) * CDIM + h * DHEAD);
#pragma unroll
    for (int e = 0; e < 8; ++e) aq[e] = q8[e];
  }

  // Pass 1: score tiles -> LDS (bf16), masked beyond LTOT
  for (int jt = 0; jt < LPAD / 16; ++jt) {
    v16bf bk = {};
    if (hi == 0) {   // B layout: K = hi*8 + e, real K only < 8
      v8bf k8 = *(const v8bf*)(Kb + ((long)b * LPAD + jt * 16 + mrow) * CDIM + h * DHEAD);
#pragma unroll
      for (int e = 0; e < 8; ++e) bk[e] = k8[e];
    }
    v8f c = {};
    c = __builtin_amdgcn_wmma_f32_16x16x32_bf16(false, aq, false, bk,
                                                (short)0, c, false, false);
    const int j = jt * 16 + mrow;
#pragma unroll
    for (int r = 0; r < 8; ++r) {
      int m = r + (hi ? 8 : 0);
      float v = (j < LTOT) ? c[r] * scale : -1e30f;
      sc[m * LPAD + j] = (__bf16)v;
    }
  }
  __syncthreads();

  // Pass 2: softmax per row; lane handles (row = mrow, half = hi); merge halves
  {
    const int m = mrow;
    const int j0 = hi * (LPAD / 2), j1 = j0 + (LPAD / 2);
    float mx = -1e30f;
    for (int j = j0; j < j1; ++j) mx = fmaxf(mx, (float)sc[m * LPAD + j]);
    mx = fmaxf(mx, __shfl_xor(mx, 16, 32));
    float sum = 0.f;
    for (int j = j0; j < j1; ++j) sum += __expf((float)sc[m * LPAD + j] - mx);
    sum += __shfl_xor(sum, 16, 32);
    const float inv = 1.f / sum;
    for (int j = j0; j < j1; ++j)
      sc[m * LPAD + j] = (__bf16)(__expf((float)sc[m * LPAD + j] - mx) * inv);
  }
  __syncthreads();

  // Pass 3: out = P @ V  (K = LPAD in 35 steps of 32; N padded 8 -> 16)
  v8f acc = {};
  for (int k0 = 0; k0 < LPAD; k0 += 32) {
    v16bf ap = load_a_frag(sc + mrow * LPAD + k0, hi);   // probs from LDS (b128)
    v16bf bv = {};
    if (mrow < DHEAD) {
#pragma unroll
      for (int e = 0; e < 16; ++e)
        bv[e] = vlds[(k0 + (e < 8 ? hi * 8 + e : 16 + hi * 8 + (e - 8))) * DHEAD + mrow];
    }
    acc = __builtin_amdgcn_wmma_f32_16x16x32_bf16(false, ap, false, bv,
                                                  (short)0, acc, false, false);
  }
  if (mrow < DHEAD) {
#pragma unroll
    for (int r = 0; r < 8; ++r) {
      int m = r + (hi ? 8 : 0);
      O[((long)b * NTOK + qt * 16 + m) * CDIM + h * DHEAD + mrow] = (__bf16)acc[r];
    }
  }
}

// ---------------------------------------------------------------------------
// Epilogue: residual add + NCHW-flat store (reference's raw reshape)
//   out[(b*C + c)*N + n] = proj[b,n,c] + hs[b,n,c]
// ---------------------------------------------------------------------------
__global__ void epilogue_k(const float* __restrict__ proj, const float* __restrict__ hs,
                           float* __restrict__ out) {
  long idx = (long)blockIdx.x * blockDim.x + threadIdx.x;
  const long total = (long)BATCH * NTOK * CDIM;
  if (idx >= total) return;
  long b = idx / ((long)NTOK * CDIM);
  long rem = idx % ((long)NTOK * CDIM);
  long n = rem / CDIM, c = rem % CDIM;
  out[(b * CDIM + c) * NTOK + n] = proj[idx] + hs[idx];
}

// ---------------------------------------------------------------------------
// Host launcher
// ---------------------------------------------------------------------------
extern "C" void kernel_launch(void* const* d_in, const int* in_sizes, int n_in,
                              void* d_out, int out_size, void* d_ws, size_t ws_size,
                              hipStream_t stream) {
  const float* hs   = (const float*)d_in[0];   // [2,32,32,512] == [B,N,C]
  const float* enc  = (const float*)d_in[1];   // [2,77,768]
  const float* gn_s = (const float*)d_in[2];
  const float* gn_b = (const float*)d_in[3];
  const float* ln_s = (const float*)d_in[4];
  const float* ln_b = (const float*)d_in[5];
  const float* Wq = (const float*)d_in[6];  const float* bq = (const float*)d_in[7];
  const float* Wk = (const float*)d_in[8];  const float* bk = (const float*)d_in[9];
  const float* Wv = (const float*)d_in[10]; const float* bv = (const float*)d_in[11];
  const float* Wak = (const float*)d_in[12]; const float* bak = (const float*)d_in[13];
  const float* Wav = (const float*)d_in[14]; const float* bav = (const float*)d_in[15];
  const float* Wo = (const float*)d_in[16]; const float* bo = (const float*)d_in[17];
  float* out = (float*)d_out;

  // ---- carve workspace ----
  char* p = (char*)d_ws;
  auto alloc = [&](size_t bytes) -> void* {
    void* r = (void*)p;
    p += (bytes + 255) & ~(size_t)255;
    return r;
  };
  __bf16* enc_bf  = (__bf16*)alloc((size_t)BATCH * SENC_P * EDIM * 2);
  __bf16* hsn_bf  = (__bf16*)alloc((size_t)BATCH * NTOK * CDIM * 2);
  __bf16* q_bf    = (__bf16*)alloc((size_t)BATCH * NTOK * CDIM * 2);
  __bf16* k_bf    = (__bf16*)alloc((size_t)BATCH * LPAD * CDIM * 2);
  __bf16* v_bf    = (__bf16*)alloc((size_t)BATCH * LPAD * CDIM * 2);
  __bf16* attn_bf = (__bf16*)alloc((size_t)BATCH * NTOK * CDIM * 2);
  float*  proj_f  = (float*) alloc((size_t)BATCH * NTOK * CDIM * 4);
  __bf16* Wq_t  = (__bf16*)alloc((size_t)CDIM * CDIM * 2);
  __bf16* Wk_t  = (__bf16*)alloc((size_t)CDIM * CDIM * 2);
  __bf16* Wv_t  = (__bf16*)alloc((size_t)CDIM * CDIM * 2);
  __bf16* Wo_t  = (__bf16*)alloc((size_t)CDIM * CDIM * 2);
  __bf16* Wak_t = (__bf16*)alloc((size_t)EDIM * CDIM * 2);
  __bf16* Wav_t = (__bf16*)alloc((size_t)EDIM * CDIM * 2);

  // ---- normalizations (fp32 stats, bf16 outputs for WMMA) ----
  layernorm_bf16_k<<<dim3(SENC_P, BATCH), 256, 0, stream>>>(enc, ln_s, ln_b, enc_bf);
  groupnorm_bf16_k<<<dim3(GROUPS, BATCH), 256, 0, stream>>>(hs, gn_s, gn_b, hsn_bf);

  // ---- weight conversion (transposed bf16) ----
  {
    int n512 = CDIM * CDIM, n768 = EDIM * CDIM;
    convert_w_t_k<<<(n512 + 255) / 256, 256, 0, stream>>>(Wq, Wq_t, CDIM, CDIM);
    convert_w_t_k<<<(n512 + 255) / 256, 256, 0, stream>>>(Wk, Wk_t, CDIM, CDIM);
    convert_w_t_k<<<(n512 + 255) / 256, 256, 0, stream>>>(Wv, Wv_t, CDIM, CDIM);
    convert_w_t_k<<<(n512 + 255) / 256, 256, 0, stream>>>(Wo, Wo_t, CDIM, CDIM);
    convert_w_t_k<<<(n768 + 255) / 256, 256, 0, stream>>>(Wak, Wak_t, EDIM, CDIM);
    convert_w_t_k<<<(n768 + 255) / 256, 256, 0, stream>>>(Wav, Wav_t, EDIM, CDIM);
  }

  // ---- zero K/V (padding rows beyond 1101 must be zero every call) ----
  {
    long nkv = (long)BATCH * LPAD * CDIM;
    zero_bf16_k<<<(int)((nkv + 255) / 256), 256, 0, stream>>>(k_bf, nkv);
    zero_bf16_k<<<(int)((nkv + 255) / 256), 256, 0, stream>>>(v_bf, nkv);
  }

  const long nBatchNC = (long)NTOK * CDIM;
  const long kvBatch  = (long)LPAD * CDIM;
  const long encBatch = (long)SENC_P * EDIM;

  // ---- projections (bf16 WMMA, 16x64 per wave) ----
  dim3 gSelf(CDIM / 64, NTOK / 16, BATCH);
  gemm_bf16_wmma_k<<<gSelf, 32, 0, stream>>>(hsn_bf, CDIM, nBatchNC, Wq_t, CDIM,
                                             bq, nullptr, q_bf, CDIM, nBatchNC, 0, NTOK);
  gemm_bf16_wmma_k<<<gSelf, 32, 0, stream>>>(hsn_bf, CDIM, nBatchNC, Wk_t, CDIM,
                                             bk, nullptr, k_bf, CDIM, kvBatch, SENC, NTOK);
  gemm_bf16_wmma_k<<<gSelf, 32, 0, stream>>>(hsn_bf, CDIM, nBatchNC, Wv_t, CDIM,
                                             bv, nullptr, v_bf, CDIM, kvBatch, SENC, NTOK);
  dim3 gEnc(CDIM / 64, SENC_P / 16, BATCH);
  gemm_bf16_wmma_k<<<gEnc, 32, 0, stream>>>(enc_bf, EDIM, encBatch, Wak_t, EDIM,
                                            bak, nullptr, k_bf, CDIM, kvBatch, 0, SENC);
  gemm_bf16_wmma_k<<<gEnc, 32, 0, stream>>>(enc_bf, EDIM, encBatch, Wav_t, EDIM,
                                            bav, nullptr, v_bf, CDIM, kvBatch, 0, SENC);

  // ---- attention (bf16 WMMA, LDS softmax + LDS-staged V) ----
  attn_wmma_k<<<dim3(NTOK / 16, HEADS, BATCH), 32, 0, stream>>>(q_bf, k_bf, v_bf, attn_bf);

  // ---- output projection (f32 result) ----
  gemm_bf16_wmma_k<<<gSelf, 32, 0, stream>>>(attn_bf, CDIM, nBatchNC, Wo_t, CDIM,
                                             bo, proj_f, nullptr, CDIM, nBatchNC, 0, NTOK);

  // ---- residual + NCHW-flat store ----
  {
    long total = (long)BATCH * NTOK * CDIM;
    epilogue_k<<<(int)((total + 255) / 256), 256, 0, stream>>>(proj_f, hs, out);
  }
}
